// DARTSCell_45715631899037
// MI455X (gfx1250) — matrix operand order repaired and checked by reading, not compiled, not run on hardware
//
#include <hip/hip_runtime.h>
#include <hip/hip_bf16.h>

typedef __attribute__((ext_vector_type(16))) __bf16 v16bf;
typedef __attribute__((ext_vector_type(8)))  __bf16 v8bf;
typedef __attribute__((ext_vector_type(8)))  float  v8f;

#define T_STEPS 256
#define BATCH   64
#define NINP    1024
#define NHID    1024
#define NBLK    32
#define THREADS 256
#define PHASES  9          // 1 (W0 phase) + 8 genotype phases per time step

#if __has_builtin(__builtin_amdgcn_sched_barrier)
#define SCHED_FENCE() __builtin_amdgcn_sched_barrier(0)
#else
#define SCHED_FENCE() asm volatile("" ::: "memory")
#endif

__device__ __forceinline__ float sigf(float x) { return 1.0f / (1.0f + __expf(-x)); }

__device__ __forceinline__ float actf(int a, float x) {
  switch (a) {
    case 0:  return sigf(x);          // sigmoid
    case 1:  return fmaxf(x, 0.0f);   // relu
    case 2:  return tanhf(x);         // tanh
    default: return x;                // identity
  }
}

__device__ __forceinline__ v8f vzero8() {
  v8f z = {0.f, 0.f, 0.f, 0.f, 0.f, 0.f, 0.f, 0.f};
  return z;
}

// Device-scope phase barrier across the 32 persistent workgroups.
// One fresh counter per phase (reset every launch) -> deterministic, replay-safe.
// Also issues the CDNA5 cluster split-barrier (NOP when not dispatched as a
// cluster; real synchronization is provided by the L2 atomic counter).
__device__ __forceinline__ void grid_barrier(unsigned* cnt, int idx) {
  __syncthreads();
  __builtin_amdgcn_fence(__ATOMIC_RELEASE, "agent");
  if (threadIdx.x == 0)
    __hip_atomic_fetch_add(&cnt[idx], 1u, __ATOMIC_RELAXED, __HIP_MEMORY_SCOPE_AGENT);
  __builtin_amdgcn_s_cluster_barrier();   // s_barrier_signal -3 ; s_barrier_wait -3
  if (threadIdx.x == 0) {
    while (__hip_atomic_load(&cnt[idx], __ATOMIC_RELAXED, __HIP_MEMORY_SCOPE_AGENT) <
           (unsigned)NBLK)
      __builtin_amdgcn_s_sleep(1);
  }
  __syncthreads();
  __builtin_amdgcn_fence(__ATOMIC_ACQUIRE, "agent");
}

union Frag { v16bf v; v8bf h[2]; };

// Software-pipelined (depth 3), fully-unrolled K-reduction: one A row-tile
// fragment feeding two B column-tile fragments (c and h halves of the fused
// weight matrix). Loads run (PIPE-1) iterations ahead of the WMMAs and the
// issue order is pinned with sched_barrier(0), so the wait inserter emits
// relaxed s_wait_loadcnt and the math overlaps L2 latency.
template <int K>
__device__ __forceinline__ void gemm_cols2(const __bf16* arow,
                                           const __bf16* wc,
                                           const __bf16* wh,
                                           int koff, int kboff, v8f& aC, v8f& aH) {
  constexpr int PIPE = 3;
  Frag a[PIPE], bc[PIPE], bh[PIPE];
#pragma unroll
  for (int p = 0; p < PIPE - 1; ++p) {     // prologue: stages 0..PIPE-2
    const int kp = p * 32;
    a[p].h[0]  = *(const v8bf*)(arow + kp + koff);
    a[p].h[1]  = *(const v8bf*)(arow + kp + 8 + koff);
    bc[p].h[0] = *(const v8bf*)(wc + kp + kboff);
    bc[p].h[1] = *(const v8bf*)(wc + kp + kboff + 8);
    bh[p].h[0] = *(const v8bf*)(wh + kp + kboff);
    bh[p].h[1] = *(const v8bf*)(wh + kp + kboff + 8);
  }
  SCHED_FENCE();
#pragma unroll
  for (int k0 = 0; k0 < K; k0 += 32) {
    const int it  = k0 >> 5;
    const int cur = it % PIPE;
    const int nxt = (it + PIPE - 1) % PIPE;
    const int kn  = k0 + (PIPE - 1) * 32;
    if (kn < K) {                          // issue loads PIPE-1 iterations ahead
      a[nxt].h[0]  = *(const v8bf*)(arow + kn + koff);
      a[nxt].h[1]  = *(const v8bf*)(arow + kn + 8 + koff);
      bc[nxt].h[0] = *(const v8bf*)(wc + kn + kboff);
      bc[nxt].h[1] = *(const v8bf*)(wc + kn + kboff + 8);
      bh[nxt].h[0] = *(const v8bf*)(wh + kn + kboff);
      bh[nxt].h[1] = *(const v8bf*)(wh + kn + kboff + 8);
    }
    SCHED_FENCE();
    aC = __builtin_amdgcn_wmma_f32_16x16x32_bf16(false, a[cur].v, false, bc[cur].v,
                                                 (short)0, aC, false, false);
    aH = __builtin_amdgcn_wmma_f32_16x16x32_bf16(false, a[cur].v, false, bh[cur].v,
                                                 (short)0, aH, false, false);
    SCHED_FENCE();
  }
}

// ---------------- conversion / init kernels ----------------

__global__ void cvt_f32_bf16(const float* __restrict__ src, __bf16* __restrict__ dst, int n) {
  for (int i = blockIdx.x * blockDim.x + threadIdx.x; i < n; i += gridDim.x * blockDim.x)
    dst[i] = (__bf16)src[i];
}

// W0 [2048 x 2048] (K-major) -> W0t [n=2048][k=2048] bf16
__global__ void cvt_w0t(const float* __restrict__ w, __bf16* __restrict__ wt) {
  for (int i = blockIdx.x * blockDim.x + threadIdx.x; i < 2048 * 2048;
       i += gridDim.x * blockDim.x) {
    int n = i >> 11, k = i & 2047;
    wt[i] = (__bf16)w[(size_t)k * 2048 + n];
  }
}

// Ws [8][1024 x 2048] -> Wst [8][n=2048][k=1024] bf16
__global__ void cvt_wst(const float* __restrict__ w, __bf16* __restrict__ wt) {
  for (int i = blockIdx.x * blockDim.x + threadIdx.x; i < 8 * 2048 * 1024;
       i += gridDim.x * blockDim.x) {
    int s = i >> 21;
    int r = i & ((1 << 21) - 1);
    int n = r >> 10, k = r & 1023;
    wt[i] = (__bf16)w[(size_t)s * 1024 * 2048 + (size_t)k * 2048 + n];
  }
}

__global__ void init_state(const float* __restrict__ h0, float* __restrict__ Hf,
                           __bf16* __restrict__ Hb, unsigned* __restrict__ cnt) {
  for (int i = blockIdx.x * blockDim.x + threadIdx.x; i < BATCH * NHID;
       i += gridDim.x * blockDim.x) {
    float v = h0[i];
    Hf[i] = v;
    Hb[i] = (__bf16)v;
  }
  for (int j = blockIdx.x * blockDim.x + threadIdx.x; j < T_STEPS * PHASES;
       j += gridDim.x * blockDim.x)
    cnt[j] = 0u;
}

// ---------------- persistent recurrent kernel ----------------

__global__ __launch_bounds__(THREADS, 1)
void darts_persist(const __bf16* __restrict__ Xb,    // [T][B][NINP] bf16
                   const __bf16* __restrict__ W0t,   // [2048][2048] bf16 (n-major)
                   const __bf16* __restrict__ Wst,   // [8][2048][1024] bf16 (n-major)
                   float*  __restrict__ stF,         // [9][B][NHID] f32 states
                   __bf16* __restrict__ stB,         // [9][B][NHID] bf16 states
                   float*  __restrict__ Hf,          // [B][NHID] f32 carry
                   __bf16* __restrict__ Hb,          // [B][NHID] bf16 carry
                   float*  __restrict__ out,         // [T][B][NHID] f32
                   unsigned* __restrict__ bar) {
  const int lane  = threadIdx.x & 31;
  const int wave  = blockIdx.x * (THREADS / 32) + (threadIdx.x >> 5);  // 0..255
  const int m0    = (wave & 3)  << 4;   // row-tile of 64-row batch
  const int n0    = (wave >> 2) << 4;   // col-tile of 1024-wide state
  const int rlane = lane & 15;
  const int koff  = (lane < 16) ? 0 : 16;   // A fragment half-group offset
  const int kboff = (lane < 16) ? 0 : 16;   // B fragment K offset
  const int row   = m0 + rlane;             // A row this lane loads
  const int col   = n0 + rlane;             // output column this lane owns
  const int rbase = (lane >= 16) ? 8 : 0;   // C/D row offset per lane half

  const __bf16* hrow = Hb + (size_t)row * NHID;

  // This lane's 8 output element indices (row-major into [B][NHID] buffers).
  size_t eidx[8];
#pragma unroll
  for (int r = 0; r < 8; ++r) eidx[r] = (size_t)(m0 + rbase + r) * NHID + col;

  for (int t = 0; t < T_STEPS; ++t) {
    // Opaque zero offset, refreshed every time step: blocks LICM from hoisting
    // the (t-invariant) weight fragment loads out of the time loop (which
    // previously spilled ~4KB of fragments to scratch), while PRESERVING the
    // global address-space provenance of the pointers so the fragment loads
    // stay global_load_b128 (pure LOADcnt) instead of flat_load_b128
    // (LOADcnt+DScnt). Weights are re-streamed from the 192MB L2 every phase.
    size_t opq = 0;
    asm volatile("" : "+s"(opq));
    const __bf16* w0c = W0t + opq + (size_t)col * 2048;
    const __bf16* w0h = W0t + opq + (size_t)(col + NHID) * 2048;
    const __bf16* wsb = Wst + opq + (size_t)col * NHID;

    // ---- phase 0: s0 = h + sigmoid(c0) * (tanh(h0) - h), A = [x_t | h] ----
    // Prefetch previous-hidden f32 values so they overlap the GEMM.
    float hp[8];
#pragma unroll
    for (int r = 0; r < 8; ++r) hp[r] = Hf[eidx[r]];

    v8f accC = vzero8(), accH = vzero8();
    const __bf16* xrow = Xb + ((size_t)t * BATCH + row) * NINP;
    // K split: first 1024 from x_t, second 1024 from h_prev (no per-iter selects)
    gemm_cols2<NINP>(xrow, w0c, w0h, koff, kboff, accC, accH);
    gemm_cols2<NHID>(hrow, w0c + NINP, w0h + NINP, koff, kboff, accC, accH);

#pragma unroll
    for (int r = 0; r < 8; ++r) {
      const float s = fmaf(sigf(accC[r]), tanhf(accH[r]) - hp[r], hp[r]);
      stF[eidx[r]] = s;
      stB[eidx[r]] = (__bf16)s;
    }
    grid_barrier(bar, t * PHASES + 0);

    // ---- phases 1..8: DARTS genotype DAG ----
    float macc[8] = {0.f, 0.f, 0.f, 0.f, 0.f, 0.f, 0.f, 0.f};
#pragma unroll
    for (int i = 0; i < 8; ++i) {
      const int PRED[8] = {0, 1, 1, 1, 2, 5, 3, 5};
      const int ACTI[8] = {0, 1, 1, 3, 2, 0, 2, 1};   // sig,relu,relu,id,tanh,sig,tanh,relu
      const int pred = PRED[i];
      const int act  = ACTI[i];
      const __bf16* arow = stB + (size_t)pred * (BATCH * NHID) + (size_t)row * NHID;
      const __bf16* wc   = wsb + (size_t)i * (2 * NHID * NHID);
      const __bf16* wh   = wc + (size_t)NHID * NHID;

      // Prefetch predecessor f32 state (overlaps the GEMM below).
      const float* spF = stF + (size_t)pred * (BATCH * NHID);
      float sp[8];
#pragma unroll
      for (int r = 0; r < 8; ++r) sp[r] = spF[eidx[r]];

      v8f aC = vzero8(), aH = vzero8();
      gemm_cols2<NHID>(arow, wc, wh, koff, kboff, aC, aH);

      float*  sF = stF + (size_t)(i + 1) * (BATCH * NHID);
      __bf16* sB = stB + (size_t)(i + 1) * (BATCH * NHID);
#pragma unroll
      for (int r = 0; r < 8; ++r) {
        const float s = fmaf(sigf(aC[r]), actf(act, aH[r]) - sp[r], sp[r]);
        sF[eidx[r]] = s;
        sB[eidx[r]] = (__bf16)s;
        macc[r] += s;
        if (i == 7) {   // h_new = mean(states 1..8); store + carry for next t
          const float hn = macc[r] * 0.125f;
          out[(size_t)t * (BATCH * NHID) + eidx[r]] = hn;
          Hf[eidx[r]] = hn;
          Hb[eidx[r]] = (__bf16)hn;
        }
      }
      grid_barrier(bar, t * PHASES + 1 + i);
    }
  }
}

// ---------------- host launcher ----------------

extern "C" void kernel_launch(void* const* d_in, const int* in_sizes, int n_in,
                              void* d_out, int out_size, void* d_ws, size_t ws_size,
                              hipStream_t stream) {
  const float* X  = (const float*)d_in[0];  // [T,B,NINP] f32
  const float* H0 = (const float*)d_in[1];  // [1,B,NHID] f32
  const float* W0 = (const float*)d_in[2];  // [2048,2048] f32
  const float* Ws = (const float*)d_in[3];  // [8,1024,2048] f32
  (void)in_sizes; (void)n_in; (void)out_size; (void)ws_size;

  char* ws = (char*)d_ws;
  size_t off = 0;
  auto alloc = [&](size_t bytes) -> char* {
    char* p = ws + off;
    off = (off + bytes + 255) & ~(size_t)255;
    return p;
  };
  unsigned* bar = (unsigned*)alloc((size_t)T_STEPS * PHASES * sizeof(unsigned));
  __bf16* Xb  = (__bf16*)alloc((size_t)T_STEPS * BATCH * NINP * 2);
  __bf16* W0t = (__bf16*)alloc((size_t)2048 * 2048 * 2);
  __bf16* Wst = (__bf16*)alloc((size_t)8 * 2048 * 1024 * 2);
  float*  stF = (float*) alloc((size_t)9 * BATCH * NHID * 4);
  __bf16* stB = (__bf16*)alloc((size_t)9 * BATCH * NHID * 2);
  float*  Hf  = (float*) alloc((size_t)BATCH * NHID * 4);
  __bf16* Hb  = (__bf16*)alloc((size_t)BATCH * NHID * 2);

  cvt_f32_bf16<<<2048, 256, 0, stream>>>(X, Xb, T_STEPS * BATCH * NINP);
  cvt_w0t     <<<1024, 256, 0, stream>>>(W0, W0t);
  cvt_wst     <<<2048, 256, 0, stream>>>(Ws, Wst);
  init_state  <<<256,  256, 0, stream>>>(H0, Hf, Hb, bar);

  darts_persist<<<NBLK, THREADS, 0, stream>>>(Xb, W0t, Wst, stF, stB, Hf, Hb,
                                              (float*)d_out, bar);
}
